// RichReLUTranscoder_28355374088378
// MI455X (gfx1250) — matrix operand order repaired and checked by reading, not compiled, not run on hardware
//
#include <hip/hip_runtime.h>

// ---------------------------------------------------------------------------
// Types for CDNA5 WMMA (wave32)
// ---------------------------------------------------------------------------
typedef __bf16 bf16_t;
typedef __attribute__((ext_vector_type(16))) __bf16 v16bf;
typedef __attribute__((ext_vector_type(8)))  float  v8f;
typedef __attribute__((ext_vector_type(4)))  unsigned int u32x4;

union FragU {
    v16bf v;
    u32x4 q[2];
};
static_assert(sizeof(FragU) == 32, "frag size");

#define B_DIM 2048
#define D_DIM 768
#define H_DIM 3072
#define L_DIM 32768
#define K_TOP 32

// LDS staging geometry (block tile 256M x 128N, K-step 32, double buffered)
#define A_ROW_BYTES 80     // 32 bf16 = 64B, padded to 80B (20 dwords) -> conflict-free frags
#define B_ROW_BYTES 272    // 128 bf16 = 256B, padded to 272B (68 dwords)
#define A_BYTES (256 * A_ROW_BYTES)   // 20480
#define B_BYTES (32 * B_ROW_BYTES)    //  8704
#define STAGE_BYTES (A_BYTES + B_BYTES)

// ---------------------------------------------------------------------------
// gfx1250 async copy global->LDS (ASYNCcnt tracked).
// LDS generic pointers carry the LDS byte offset in their low 32 bits
// ---------------------------------------------------------------------------
__device__ __forceinline__ void rrt_async_b128(void* lds_ptr, const void* gptr) {
    unsigned loff = (unsigned)(size_t)lds_ptr;
    asm volatile("global_load_async_to_lds_b128 %0, %1, off"
                 :: "v"(loff), "v"(gptr)
                 : "memory");
}

// ---------------------------------------------------------------------------
// f32 -> bf16 conversion (grid-stride, bandwidth bound)
// ---------------------------------------------------------------------------
__global__ void rrt_cvt_bf16(const float* __restrict__ src,
                             bf16_t* __restrict__ dst, size_t n) {
    size_t i = (size_t)blockIdx.x * blockDim.x + threadIdx.x;
    size_t stride = (size_t)gridDim.x * blockDim.x;
    for (; i < n; i += stride) {
        dst[i] = (bf16_t)src[i];
    }
}

// ---------------------------------------------------------------------------
// bf16 WMMA GEMM with async-LDS double buffering.
//   C[M,N] (f32) = A[M,K] @ B[K,N], optional ReLU, optional bf16 copy of C.
// Block: 256 threads = 8 waves arranged 4(M) x 2(N).
// Wave tile: 64x64 = 4x4 accumulators of 16x16 -> 16 v_wmma per K-step.
// Block tile: 256(M) x 128(N); K-step 32.
// Per K-step per thread: 6 async B128 copies; per wave: 8 ds_load_b128 (A) + 64 2-byte B loads.
// All dims divide evenly -> EXEC all ones (WMMA requirement).
// ---------------------------------------------------------------------------
__global__ __launch_bounds__(256)
void rrt_wmma_gemm_bf16(const bf16_t* __restrict__ A,
                        const bf16_t* __restrict__ Bm,
                        float* __restrict__ C,
                        bf16_t* __restrict__ Cb,
                        int M, int N, int K, int relu) {
    __shared__ alignas(16) unsigned char smem[2 * STAGE_BYTES];

    const int lane  = threadIdx.x & 31;
    const int wave  = threadIdx.x >> 5;
    const int waveM = wave >> 1;   // 0..3
    const int waveN = wave & 1;    // 0..1
    const int half  = lane >> 4;   // 0/1
    const int l15   = lane & 15;

    const int mBlock = blockIdx.y * 256;
    const int nBlock = blockIdx.x * 128;
    const int mTile  = mBlock + waveM * 64;
    const int nTile  = nBlock + waveN * 64;

    // ---- async stage issue: copy A(256x32) and B(32x128) tiles to LDS ----
    auto issue_stage = [&](int kstep, int buf) {
        const unsigned baseA = (unsigned)buf * STAGE_BYTES;
        const unsigned baseB = baseA + A_BYTES;
        const int k0 = kstep * 32;
        // A tile: 1024 chunks of 16B (4 per thread)
#pragma unroll
        for (int c = 0; c < 4; ++c) {
            int ch  = (int)threadIdx.x + c * 256;
            int row = ch >> 2;
            int cc  = ch & 3;              // 16B chunk within 64B row
            rrt_async_b128(smem + baseA + row * A_ROW_BYTES + cc * 16,
                           A + (size_t)(mBlock + row) * K + k0 + cc * 8);
        }
        // B tile: 512 chunks of 16B (2 per thread)
#pragma unroll
        for (int c = 0; c < 2; ++c) {
            int ch  = (int)threadIdx.x + c * 256;
            int row = ch >> 4;
            int cc  = ch & 15;             // 16B chunk within 256B row
            rrt_async_b128(smem + baseB + row * B_ROW_BYTES + cc * 16,
                           Bm + (size_t)(k0 + row) * N + nBlock + cc * 8);
        }
    };

    v8f acc[4][4];
#pragma unroll
    for (int i = 0; i < 4; i++)
#pragma unroll
        for (int j = 0; j < 4; j++)
#pragma unroll
            for (int r = 0; r < 8; r++)
                acc[i][j][r] = 0.0f;

    const int Ksteps = K / 32;
    issue_stage(0, 0);

    for (int s = 0; s < Ksteps; ++s) {
        const int cur = s & 1;
        if (s + 1 < Ksteps) {
            issue_stage(s + 1, (s + 1) & 1);
            // our stage-s copies (6 in-order async ops) retired when <= 6 remain
            asm volatile("s_wait_asynccnt 0x6" ::: "memory");
        } else {
            asm volatile("s_wait_asynccnt 0x0" ::: "memory");
        }
        __syncthreads();   // everyone's stage-s copies now visible in LDS

        const unsigned char* bufA = smem + cur * STAGE_BYTES;
        const unsigned char* bufB = bufA + A_BYTES;

        // A fragments (16-bit A 16x32 layout): lane m=l15,
        //   elems 0..7 -> K = 8*half.. ; elems 8..15 -> K = 16+8*half..
        FragU fa[4];
#pragma unroll
        for (int i = 0; i < 4; i++) {
            const unsigned char* p =
                bufA + (waveM * 64 + i * 16 + l15) * A_ROW_BYTES + 16 * half;
            fa[i].q[0] = *(const u32x4*)(p);
            fa[i].q[1] = *(const u32x4*)(p + 32);
        }
        // B fragments (16-bit B 32x16 layout): lane n=l15,
        //   elems 0..7 -> K = 8*half.. ; elems 8..15 -> K = 16+8*half..
        FragU fb[4];
#pragma unroll
        for (int j = 0; j < 4; j++) {
            const unsigned char* p =
                bufB + (waveN * 64 + j * 16 + l15) * 2;
#pragma unroll
            for (int e = 0; e < 16; e++) {
                int kk = (e < 8) ? (8 * half + e) : (16 + 8 * half + (e - 8));
                fb[j].v[e] = *(const bf16_t*)(p + kk * B_ROW_BYTES);
            }
        }
#pragma unroll
        for (int i = 0; i < 4; i++)
#pragma unroll
            for (int j = 0; j < 4; j++)
                acc[i][j] = __builtin_amdgcn_wmma_f32_16x16x32_bf16(
                    false, fa[i].v, false, fb[j].v,
                    (short)0, acc[i][j], false, false);

        __syncthreads();   // stage-s buffer free for reuse at stage s+2
    }

    // Store: 32-bit C/D 16x16 layout: VGPR r -> M = r + 8*half, N = l15.
#pragma unroll
    for (int i = 0; i < 4; i++) {
#pragma unroll
        for (int j = 0; j < 4; j++) {
#pragma unroll
            for (int r = 0; r < 8; r++) {
                int row = mTile + i * 16 + half * 8 + r;
                int col = nTile + j * 16 + l15;
                float v = acc[i][j][r];
                if (relu) v = v > 0.0f ? v : 0.0f;
                C[(size_t)row * N + col] = v;
                if (Cb) Cb[(size_t)row * N + col] = (bf16_t)v;
            }
        }
    }
}

// ---------------------------------------------------------------------------
// Top-K per row.  One block (128 threads) per batch row.
// Monotone key: flip float bits so unsigned compare == float compare.
// ---------------------------------------------------------------------------
__device__ __forceinline__ unsigned rrt_mono(float f) {
    unsigned u = __float_as_uint(f);
    return (u & 0x80000000u) ? ~u : (u | 0x80000000u);
}
__device__ __forceinline__ float rrt_unmono(unsigned k) {
    unsigned u = (k & 0x80000000u) ? (k & 0x7FFFFFFFu) : ~k;
    return __uint_as_float(u);
}

__global__ __launch_bounds__(128)
void rrt_topk(const float* __restrict__ pre,
              float* __restrict__ acts,
              int* __restrict__ idx_out,
              float* __restrict__ vals_ws,
              int* __restrict__ idx_ws) {
    __shared__ unsigned skey[128 * 32];
    __shared__ unsigned sidx[128 * 32];
    __shared__ unsigned long long red[128];

    const int b = blockIdx.x;
    const int t = threadIdx.x;
    const float* __restrict__ row = pre + (size_t)b * L_DIM;
    float* __restrict__ arow = acts + (size_t)b * L_DIM;

    // zero the sparse-activation output row
    for (int i = t; i < L_DIM; i += 128) arow[i] = 0.0f;
    // make this wave's zero-stores land before any scatter (barrier covers rest)
    asm volatile("s_wait_storecnt 0x0" ::: "memory");

    // stage 1: per-thread top-32 of its 256 strided elements
    unsigned* myk = &skey[t * 32];
    unsigned* myi = &sidx[t * 32];
#pragma unroll
    for (int s = 0; s < 32; s++) myk[s] = 0u;
    unsigned kmin = 0u;
    int pmin = 0;
    for (int i = t; i < L_DIM; i += 128) {
        unsigned key = rrt_mono(row[i]);
        if (key > kmin) {
            myk[pmin] = key;
            myi[pmin] = (unsigned)i;
            kmin = myk[0]; pmin = 0;
#pragma unroll
            for (int s = 1; s < 32; s++) {
                if (myk[s] < kmin) { kmin = myk[s]; pmin = s; }
            }
        }
    }
    __syncthreads();

    // stage 2: 32 argmax rounds over the 4096 candidates (descending order)
    for (int r = 0; r < K_TOP; r++) {
        unsigned bestk = 0u;
        int bests = t * 32;
#pragma unroll
        for (int s = 0; s < 32; s++) {
            unsigned k = skey[t * 32 + s];
            if (k > bestk) { bestk = k; bests = t * 32 + s; }
        }
        red[t] = ((unsigned long long)bestk << 32) | (unsigned)bests;
        __syncthreads();
        for (int h = 64; h > 0; h >>= 1) {
            if (t < h) {
                unsigned long long o = red[t + h];
                if (o > red[t]) red[t] = o;
            }
            __syncthreads();
        }
        if (t == 0) {
            unsigned long long win = red[0];
            unsigned slot = (unsigned)(win & 0xFFFFFFFFu);
            unsigned key  = (unsigned)(win >> 32);
            float v = rrt_unmono(key);
            int gi = (int)sidx[slot];
            idx_out[b * K_TOP + r] = gi;
            vals_ws[b * K_TOP + r] = v;
            idx_ws[b * K_TOP + r]  = gi;
            arow[gi] = v;
            skey[slot] = 0u;   // exclude winner
        }
        __syncthreads();
    }
}

// ---------------------------------------------------------------------------
// Sparse decoder: recon[b, :] = sum_k vals[b,k] * dec[idx[b,k], :]
// ---------------------------------------------------------------------------
__global__ __launch_bounds__(256)
void rrt_decoder(const float* __restrict__ vals,
                 const int* __restrict__ idxs,
                 const float* __restrict__ dec,
                 float* __restrict__ recon) {
    __shared__ float sv[K_TOP];
    __shared__ int   si[K_TOP];
    const int b = blockIdx.x;
    if (threadIdx.x < K_TOP) {
        sv[threadIdx.x] = vals[b * K_TOP + threadIdx.x];
        si[threadIdx.x] = idxs[b * K_TOP + threadIdx.x];
    }
    __syncthreads();
    for (int d = threadIdx.x; d < D_DIM; d += 256) {
        float acc = 0.0f;
#pragma unroll 8
        for (int k = 0; k < K_TOP; k++)
            acc += sv[k] * dec[(size_t)si[k] * D_DIM + d];
        recon[(size_t)b * D_DIM + d] = acc;
    }
}

// ---------------------------------------------------------------------------
// Launch
// ---------------------------------------------------------------------------
extern "C" void kernel_launch(void* const* d_in, const int* in_sizes, int n_in,
                              void* d_out, int out_size, void* d_ws, size_t ws_size,
                              hipStream_t stream) {
    const float* in_act = (const float*)d_in[0];  // (B, D)
    const float* wup    = (const float*)d_in[1];  // (D, H)
    const float* enc    = (const float*)d_in[2];  // (H, L)
    const float* dec    = (const float*)d_in[3];  // (L, D)

    // d_out: ff_hidden | latent_pre | latent_acts | recon | indices (int32)
    float* out_ff    = (float*)d_out;
    float* out_pre   = out_ff   + (size_t)B_DIM * H_DIM;
    float* out_acts  = out_pre  + (size_t)B_DIM * L_DIM;
    float* out_recon = out_acts + (size_t)B_DIM * L_DIM;
    int*   out_idx   = (int*)(out_recon + (size_t)B_DIM * D_DIM);

    // workspace carve-up (256B aligned)
    char* w = (char*)d_ws;
    auto carve = [&](size_t bytes) {
        char* p = w;
        w += (bytes + 255) & ~(size_t)255;
        return p;
    };
    bf16_t* in_b  = (bf16_t*)carve((size_t)B_DIM * D_DIM * 2);
    bf16_t* wup_b = (bf16_t*)carve((size_t)D_DIM * H_DIM * 2);
    bf16_t* enc_b = (bf16_t*)carve((size_t)H_DIM * L_DIM * 2);
    bf16_t* ff_b  = (bf16_t*)carve((size_t)B_DIM * H_DIM * 2);
    float*  vals_ws = (float*)carve((size_t)B_DIM * K_TOP * 4);
    int*    idx_ws  = (int*)carve((size_t)B_DIM * K_TOP * 4);

    // 1. Convert GEMM operands to bf16 once (bandwidth bound)
    rrt_cvt_bf16<<<2048, 256, 0, stream>>>(in_act, in_b, (size_t)B_DIM * D_DIM);
    rrt_cvt_bf16<<<2048, 256, 0, stream>>>(wup, wup_b, (size_t)D_DIM * H_DIM);
    rrt_cvt_bf16<<<8192, 256, 0, stream>>>(enc, enc_b, (size_t)H_DIM * L_DIM);

    // 2. GEMM1: ff = relu(in @ W_up); f32 output + bf16 copy for GEMM2
    rrt_wmma_gemm_bf16<<<dim3(H_DIM / 128, B_DIM / 256), 256, 0, stream>>>(
        in_b, wup_b, out_ff, ff_b, B_DIM, H_DIM, D_DIM, 1);

    // 3. GEMM2 (dominant, 412 GFLOP): latent_pre = ff @ enc
    rrt_wmma_gemm_bf16<<<dim3(L_DIM / 128, B_DIM / 256), 256, 0, stream>>>(
        ff_b, enc_b, out_pre, (bf16_t*)nullptr, B_DIM, L_DIM, H_DIM, 0);

    // 4. Top-K (+ zero & scatter latent_acts, emit indices)
    rrt_topk<<<B_DIM, 128, 0, stream>>>(out_pre, out_acts, out_idx, vals_ws, idx_ws);

    // 5. Sparse decode
    rrt_decoder<<<B_DIM, 256, 0, stream>>>(vals_ws, idx_ws, dec, out_recon);
}
